// RGCN_29996051595992
// MI455X (gfx1250) — matrix operand order repaired and checked by reading, not compiled, not run on hardware
//
#include <hip/hip_runtime.h>

// Problem constants (from reference)
#define G_N 5000
#define T_N 365
#define F_N 20
#define H_N 64
// Padded GEMM dims: M padded to 16 (WMMA M), K padded to 32 (WMMA K for bf16)
#define MP 5008
#define KP 5024
#define NTILES (MP / 16)    // 313 row tiles of the big GEMM
#define NKSTEPS (KP / 32)   // 157 K iterations

typedef __attribute__((ext_vector_type(16))) __bf16 v16bf;
typedef __attribute__((ext_vector_type(8)))  __bf16 v8bf;
typedef __attribute__((ext_vector_type(8)))  float  v8f;
typedef __attribute__((ext_vector_type(4)))  int    v4i;

#if __has_builtin(__builtin_amdgcn_global_load_async_to_lds_b128) && \
    __has_builtin(__builtin_amdgcn_s_wait_asynccnt)
#define USE_ASYNC_LDS 1
#else
#define USE_ASYNC_LDS 0
#endif

__device__ __forceinline__ float sigmoidf_(float x) {
    return 1.0f / (1.0f + __expf(-x));
}

// Stage one 16B chunk global -> LDS (async on gfx1250, sync fallback otherwise)
__device__ __forceinline__ void stage_chunk(const __bf16* gsrc, __bf16* ldst) {
#if USE_ASYNC_LDS
    __builtin_amdgcn_global_load_async_to_lds_b128(
        (v4i*)gsrc,
        (__attribute__((address_space(3))) v4i*)ldst,
        0, 0);
#else
    *(v8bf*)ldst = *(const v8bf*)gsrc;
#endif
}

template <int N>
__device__ __forceinline__ void wait_async() {
#if USE_ASYNC_LDS
    __builtin_amdgcn_s_wait_asynccnt(N);
#endif
}

// ---------------------------------------------------------------------------
// Zero h and c state (contiguous region of n floats)
// ---------------------------------------------------------------------------
__global__ void k_zero_state(float* p, int n) {
    int i = blockIdx.x * blockDim.x + threadIdx.x;
    int stride = gridDim.x * blockDim.x;
    for (; i < n; i += stride) p[i] = 0.0f;
}

// ---------------------------------------------------------------------------
// One-time: A [G,G] f32 -> padded bf16 A_bf [MP,KP] (pad = 0)
// ---------------------------------------------------------------------------
__global__ void k_convert_A(const float* __restrict__ A, __bf16* __restrict__ Abf) {
    size_t idx = (size_t)blockIdx.x * blockDim.x + threadIdx.x;
    size_t total = (size_t)MP * KP;
    size_t stride = (size_t)gridDim.x * blockDim.x;
    for (; idx < total; idx += stride) {
        int r = (int)(idx / KP);
        int c = (int)(idx % KP);
        float v = (r < G_N && c < G_N) ? A[(size_t)r * G_N + c] : 0.0f;
        Abf[idx] = (__bf16)v;
    }
}

// ---------------------------------------------------------------------------
// Per step: Yt[n][m] = bf16( c_prev[m,:] @ W_graph_c[:,n] + b_graph_c[n] )
// Stored transposed [H_N, KP] so WMMA B-fragments read contiguous K.
// Pad rows m >= G_N are written as 0.
// ---------------------------------------------------------------------------
__global__ void k_y_panel(const float* __restrict__ c,
                          const float* __restrict__ Wg,
                          const float* __restrict__ bg,
                          __bf16* __restrict__ Yt) {
    int idx = blockIdx.x * blockDim.x + threadIdx.x;   // exactly H_N*KP threads
    int n = idx / KP;
    int m = idx % KP;
    float acc = 0.0f;
    if (m < G_N) {
        acc = bg[n];
        const float* crow = c + (size_t)m * H_N;
        #pragma unroll 16
        for (int k = 0; k < H_N; ++k) acc += crow[k] * Wg[k * H_N + n];
    }
    Yt[(size_t)n * KP + m] = (__bf16)acc;
}

// ---------------------------------------------------------------------------
// Big GEMM: c_graph[g][n] = tanh( sum_m A_bf[g][m] * Yt[n][m] )
// Wave32 WMMA bf16 16x16x32, f32 accumulate.
// Block = 4 waves, each wave a 16-row M-tile; the shared 64x32 B slab is
// double-buffered in LDS via async global->LDS copies (ASYNCcnt).
// ---------------------------------------------------------------------------
#define SHUF16(lo, hi) __builtin_shufflevector(lo, hi, 0,1,2,3,4,5,6,7,8,9,10,11,12,13,14,15)

__global__ void __launch_bounds__(128)
k_graph_gemm(const __bf16* __restrict__ Abf,
             const __bf16* __restrict__ Yt,
             float* __restrict__ cg) {
    __shared__ __attribute__((aligned(16))) __bf16 sB[2][H_N][32];   // 2 x 4KB

    const int lane = threadIdx.x & 31;
    int tile = blockIdx.x * 4 + (threadIdx.x >> 5);
    if (tile >= NTILES) tile = NTILES - 1;   // clamp: duplicate work, keeps
                                             // barriers uniform & EXEC all-ones
    const int m0 = tile * 16;

    v8f acc0 = {}, acc1 = {}, acc2 = {}, acc3 = {};

    // A fragment addressing (contiguous k = 16h + s, matching B):
    // lanes 0-15: row M=lane, K=0..15; lanes 16-31: K=16..31
    const int arow = m0 + (lane & 15);
    const __bf16* aptr = Abf + (size_t)arow * KP + ((lane >> 4) << 4);

    // B fragment addressing (contiguous k = 16h + s, matching A):
    // lanes 0-15: col N=lane, K=0..15; lanes 16-31: col N=lane-16, K=16..31
    const int ncol  = lane & 15;
    const int kboff = (lane >> 4) << 4;

    // Staging coordinates: 64 rows x 64B = 256 x 16B chunks, 2 per thread
    const int srow = threadIdx.x >> 2;          // 0..31
    const int skel = (threadIdx.x & 3) << 3;    // element offset 0,8,16,24
    const __bf16* g0 = Yt + (size_t)srow * KP + skel;
    const __bf16* g1 = Yt + (size_t)(srow + 32) * KP + skel;

    // Prologue: stage kb=0 into buffer 0
    stage_chunk(g0, &sB[0][srow][skel]);
    stage_chunk(g1, &sB[0][srow + 32][skel]);

    for (int i = 0; i < NKSTEPS; ++i) {
        const int kb = i * 32;
        const int buf = i & 1;

        __syncthreads();   // all waves done reading sB[buf^1] before restaging
        if (i + 1 < NKSTEPS) {
            stage_chunk(g0 + kb + 32, &sB[buf ^ 1][srow][skel]);
            stage_chunk(g1 + kb + 32, &sB[buf ^ 1][srow + 32][skel]);
            wait_async<2>();   // retire current buffer's copies, next stays in flight
        } else {
            wait_async<0>();
        }
        __syncthreads();   // sB[buf] complete across all waves

        __builtin_prefetch(aptr + kb + 256, 0, 1);   // global_prefetch_b8 on A stream
        v8bf al = *(const v8bf*)(aptr + kb);
        v8bf ah = *(const v8bf*)(aptr + kb + 8);
        v16bf a = SHUF16(al, ah);

        v16bf b0 = SHUF16(*(const v8bf*)&sB[buf][ncol][kboff],
                          *(const v8bf*)&sB[buf][ncol][kboff + 8]);
        v16bf b1 = SHUF16(*(const v8bf*)&sB[buf][16 + ncol][kboff],
                          *(const v8bf*)&sB[buf][16 + ncol][kboff + 8]);
        v16bf b2 = SHUF16(*(const v8bf*)&sB[buf][32 + ncol][kboff],
                          *(const v8bf*)&sB[buf][32 + ncol][kboff + 8]);
        v16bf b3 = SHUF16(*(const v8bf*)&sB[buf][48 + ncol][kboff],
                          *(const v8bf*)&sB[buf][48 + ncol][kboff + 8]);

        acc0 = __builtin_amdgcn_wmma_f32_16x16x32_bf16(false, a, false, b0, (short)0, acc0, false, false);
        acc1 = __builtin_amdgcn_wmma_f32_16x16x32_bf16(false, a, false, b1, (short)0, acc1, false, false);
        acc2 = __builtin_amdgcn_wmma_f32_16x16x32_bf16(false, a, false, b2, (short)0, acc2, false, false);
        acc3 = __builtin_amdgcn_wmma_f32_16x16x32_bf16(false, a, false, b3, (short)0, acc3, false, false);
    }

    // C/D layout: VGPR r -> M = r (lanes 0-15) / r+8 (lanes 16-31); N = lane&15
    const int rbase = m0 + ((lane >> 4) << 3);
    #pragma unroll
    for (int r = 0; r < 8; ++r) {
        int row = rbase + r;
        if (row < G_N) {
            float* o = cg + (size_t)row * H_N + ncol;
            o[0]  = tanhf(acc0[r]);
            o[16] = tanhf(acc1[r]);
            o[32] = tanhf(acc2[r]);
            o[48] = tanhf(acc3[r]);
        }
    }
}

// ---------------------------------------------------------------------------
// Fused LSTM gates + custom cell update + output projection.
// Block = 256 threads = 4 nodes x 64 hidden units. Grid = 1250 (exact).
// ---------------------------------------------------------------------------
__global__ void __launch_bounds__(256)
k_cell(const float* __restrict__ inputs,
       const float* __restrict__ Wx,    // lstm_kernel [F,4H]
       const float* __restrict__ Wh,    // lstm_rec_kernel [H,4H]
       const float* __restrict__ bz,    // lstm_bias [4H]
       const float* __restrict__ Wcc,   // W_c_cur [H,H]
       const float* __restrict__ Wcp,   // W_c_prev [H,H]
       const float* __restrict__ bc,    // b_c [H]
       const float* __restrict__ Wout,  // W_out [H,2]
       const float* __restrict__ bout,  // b_out [2]
       const float* __restrict__ cg,    // c_graph [G,H]
       float* __restrict__ h,           // state [G,H]
       float* __restrict__ c,           // state [G,H]
       float* __restrict__ out,         // [G,T,2]
       int t) {
    __shared__ float sx[4][F_N];
    __shared__ float sh[4][H_N];
    __shared__ float scc[4][H_N];
    __shared__ float shn[4][H_N];

    const int nl = threadIdx.x >> 6;    // node within block
    const int j  = threadIdx.x & 63;    // hidden unit
    const int g  = blockIdx.x * 4 + nl; // always < G_N (1250*4 == 5000)

    if (j < F_N) sx[nl][j] = inputs[((size_t)g * T_N + t) * F_N + j];
    sh[nl][j] = h[(size_t)g * H_N + j];
    __syncthreads();

    float z[4];
    #pragma unroll
    for (int q = 0; q < 4; ++q) {
        const int col = q * H_N + j;
        float a = bz[col];
        #pragma unroll
        for (int f = 0; f < F_N; ++f) a += sx[nl][f] * Wx[f * 4 * H_N + col];
        #pragma unroll 8
        for (int k = 0; k < H_N; ++k) a += sh[nl][k] * Wh[k * 4 * H_N + col];
        z[q] = a;
    }

    const float ig = sigmoidf_(z[0]);
    const float fg = sigmoidf_(z[1]);
    const float gg = tanhf(z[2]);
    const float og = sigmoidf_(z[3]);
    const float cp = c[(size_t)g * H_N + j];
    const float ccur = fg * cp + ig * gg;
    const float hnew = og * tanhf(ccur);

    scc[nl][j] = ccur;
    shn[nl][j] = hnew;
    __syncthreads();

    float a = bc[j];
    const float* cgr = cg + (size_t)g * H_N;
    #pragma unroll 8
    for (int k = 0; k < H_N; ++k)
        a += scc[nl][k] * Wcc[k * H_N + j] + cgr[k] * Wcp[k * H_N + j];
    const float cupd = sigmoidf_(a);

    h[(size_t)g * H_N + j] = hnew;
    c[(size_t)g * H_N + j] = cupd;

    if (j < 2) {
        float o = bout[j];
        #pragma unroll 8
        for (int k = 0; k < H_N; ++k) o += shn[nl][k] * Wout[k * 2 + j];
        out[((size_t)g * T_N + t) * 2 + j] = o;
    }
}

// ---------------------------------------------------------------------------
// Launch
// ---------------------------------------------------------------------------
extern "C" void kernel_launch(void* const* d_in, const int* in_sizes, int n_in,
                              void* d_out, int out_size, void* d_ws, size_t ws_size,
                              hipStream_t stream) {
    (void)in_sizes; (void)n_in; (void)out_size; (void)ws_size;

    const float* inputs = (const float*)d_in[0];   // [G,T,F]
    const float* A      = (const float*)d_in[1];   // [G,G]
    const float* Wg     = (const float*)d_in[2];   // W_graph_c [H,H]
    const float* bg     = (const float*)d_in[3];   // b_graph_c [H]
    const float* Wcc    = (const float*)d_in[4];   // W_c_cur [H,H]
    const float* Wcp    = (const float*)d_in[5];   // W_c_prev [H,H]
    const float* bc     = (const float*)d_in[6];   // b_c [H]
    const float* Wout   = (const float*)d_in[7];   // W_out [H,2]
    const float* bout   = (const float*)d_in[8];   // b_out [2]
    const float* Wx     = (const float*)d_in[9];   // lstm_kernel [F,4H]
    const float* Wh     = (const float*)d_in[10];  // lstm_rec_kernel [H,4H]
    const float* bz     = (const float*)d_in[11];  // lstm_bias [4H]

    float* out = (float*)d_out;                    // [G,T,2] f32

    // Workspace partition (all offsets 16B aligned)
    char* ws = (char*)d_ws;
    const size_t szA = (size_t)MP * KP * sizeof(__bf16);          // 50.3 MB
    const size_t szY = (size_t)H_N * KP * sizeof(__bf16);         // 643 KB
    const size_t szS = (size_t)G_N * H_N * sizeof(float);         // 1.28 MB
    __bf16* Abf = (__bf16*)(ws);
    __bf16* Yt  = (__bf16*)(ws + szA);
    float*  h   = (float*)(ws + szA + szY);
    float*  c   = (float*)(ws + szA + szY + szS);
    float*  cg  = (float*)(ws + szA + szY + 2 * szS);

    // Init: zero h and c (contiguous), convert A to padded bf16
    k_zero_state<<<512, 256, 0, stream>>>(h, 2 * G_N * H_N);
    k_convert_A<<<4096, 256, 0, stream>>>(A, Abf);

    const int gemm_blocks = (NTILES + 3) / 4;        // 79
    const int ypanel_blocks = (H_N * KP) / 256;      // 1256 (exact)
    const int cell_blocks = G_N / 4;                 // 1250 (exact)

    for (int t = 0; t < T_N; ++t) {
        k_y_panel<<<ypanel_blocks, 256, 0, stream>>>(c, Wg, bg, Yt);
        k_graph_gemm<<<gemm_blocks, 128, 0, stream>>>(Abf, Yt, cg);
        k_cell<<<cell_blocks, 256, 0, stream>>>(inputs, Wx, Wh, bz, Wcc, Wcp, bc,
                                                Wout, bout, cg, h, c, out, t);
    }
}
